// SelfAttTransformerEncoderLayer_68959994904865
// MI455X (gfx1250) — matrix operand order repaired and checked by reading, not compiled, not run on hardware
//
#include <hip/hip_runtime.h>
#include <math.h>

// ---------------------------------------------------------------------------
// Types for CDNA5 WMMA (wave32)
// ---------------------------------------------------------------------------
typedef __attribute__((ext_vector_type(16))) __bf16 v16bf;
typedef __attribute__((ext_vector_type(8)))  __bf16 v8bf;
typedef __attribute__((ext_vector_type(4)))  __bf16 v4bf;
typedef __attribute__((ext_vector_type(8)))  float  v8f;

#define D_MODEL 256
#define NHEAD   8
#define DHEAD   32

// ---------------------------------------------------------------------------
// helpers
// ---------------------------------------------------------------------------
__device__ __forceinline__ float gelu_tanh(float x) {
    float x3 = x * x * x;
    return 0.5f * x * (1.0f + tanhf(0.7978845608028654f * (x + 0.044715f * x3)));
}

// monotonic float <-> uint mapping for atomic max on signed floats
__device__ __forceinline__ unsigned mapf(float f) {
    unsigned u = __float_as_uint(f);
    return (u & 0x80000000u) ? ~u : (u | 0x80000000u);
}
__device__ __forceinline__ float unmapf(unsigned u) {
    unsigned b = (u & 0x80000000u) ? (u & 0x7FFFFFFFu) : ~u;
    return __uint_as_float(b);
}

// A-fragment (16x32 bf16): lane m = lane%16, half = lane/16.
// elems 0..15 -> K = half*16 + e (contiguous, matches B fragment)
__device__ __forceinline__ v16bf load_a_frag(const __bf16* rowptr, int k0, int half) {
    v8bf lo = *(const v8bf*)(rowptr + k0 + half * 16);
    v8bf hi = *(const v8bf*)(rowptr + k0 + half * 16 + 8);
    return __builtin_shufflevector(lo, hi, 0,1,2,3,4,5,6,7,8,9,10,11,12,13,14,15);
}

// ---------------------------------------------------------------------------
// Weight conversion: W[K,Nout] f32 -> WT[Nout][K] bf16 (coalesced writes)
// ---------------------------------------------------------------------------
__global__ void w_to_bf16T_kernel(const float* __restrict__ W, __bf16* __restrict__ WT,
                                  int K, int Nout) {
    int gid = blockIdx.x * 256 + threadIdx.x;
    if (gid >= K * Nout) return;
    int n = gid / K;
    int k = gid - n * K;
    WT[gid] = (__bf16)W[(size_t)k * Nout + n];
}

// ---------------------------------------------------------------------------
// LayerNorm (rows of 256) -> bf16 output. One wave per row, 8 rows per block.
// ---------------------------------------------------------------------------
__global__ __launch_bounds__(256)
void ln_bf16_kernel(const float* __restrict__ X, const float* __restrict__ g,
                    const float* __restrict__ b, __bf16* __restrict__ Y, int M) {
    int wid  = threadIdx.x >> 5;
    int lane = threadIdx.x & 31;
    int row  = blockIdx.x * 8 + wid;
    if (row >= M) return;
    const float4* x4 = (const float4*)(X + (size_t)row * D_MODEL);
    float4 a0 = x4[lane];
    float4 a1 = x4[lane + 32];
    float s = a0.x + a0.y + a0.z + a0.w + a1.x + a1.y + a1.z + a1.w;
    float q = a0.x*a0.x + a0.y*a0.y + a0.z*a0.z + a0.w*a0.w
            + a1.x*a1.x + a1.y*a1.y + a1.z*a1.z + a1.w*a1.w;
    #pragma unroll
    for (int m = 16; m >= 1; m >>= 1) {
        s += __shfl_xor(s, m, 32);
        q += __shfl_xor(q, m, 32);
    }
    float mean = s * (1.0f / 256.0f);
    float var  = q * (1.0f / 256.0f) - mean * mean;
    float rstd = rsqrtf(var + 1e-5f);

    float va[8] = {a0.x, a0.y, a0.z, a0.w, a1.x, a1.y, a1.z, a1.w};
    int   c0[2] = {lane * 4, 128 + lane * 4};
    #pragma unroll
    for (int h = 0; h < 2; ++h) {
        v4bf o;
        #pragma unroll
        for (int i = 0; i < 4; ++i) {
            int c = c0[h] + i;
            o[i] = (__bf16)((va[h * 4 + i] - mean) * rstd * g[c] + b[c]);
        }
        *(v4bf*)(Y + (size_t)row * D_MODEL + c0[h]) = o;
    }
}

// ---------------------------------------------------------------------------
// Generic bf16 WMMA GEMM: C[M,Nout] = A[M,K]@W + bias (+resid) (gelu) -> f32/bf16
// Block: 256 thr (8 waves). Tile 64(M) x 128(N). LDS-staged A, K chunk = 256.
// ---------------------------------------------------------------------------
#define GEMM_FLAG_GELU 1

__global__ __launch_bounds__(256)
void gemm_bf16_kernel(const __bf16* __restrict__ A, const __bf16* __restrict__ WT,
                      const float* __restrict__ bias, const float* __restrict__ resid,
                      float* __restrict__ outF, __bf16* __restrict__ outB,
                      int M, int K, int Nout, int flags) {
    __shared__ __bf16 As[64][256 + 8];
    const int tid  = threadIdx.x;
    const int wid  = tid >> 5, lane = tid & 31;
    const int wm   = wid & 3;        // 0..3 : 16-row subtile
    const int wn   = wid >> 2;       // 0..1 : 64-col half
    const int half = lane >> 4, l16 = lane & 15;
    const int bm   = blockIdx.y * 64;
    const int bn   = blockIdx.x * 128;

    v8f acc[4] = {v8f{}, v8f{}, v8f{}, v8f{}};

    for (int kc = 0; kc < K; kc += 256) {
        __syncthreads();
        #pragma unroll
        for (int it = 0; it < 8; ++it) {
            int p = (it * 256 + tid) * 8;        // 64*256 elems, 8 per thread per it
            int r = p >> 8, c = p & 255;
            int gr = bm + r;
            v8bf val = v8bf{};
            if (gr < M) val = *(const v8bf*)(A + (size_t)gr * K + kc + c);
            *(v8bf*)(&As[r][c]) = val;
        }
        __syncthreads();
        const __bf16* arow = &As[wm * 16 + l16][0];
        #pragma unroll
        for (int k0 = 0; k0 < 256; k0 += 32) {
            v16bf af = load_a_frag(arow, k0, half);
            #pragma unroll
            for (int ns = 0; ns < 4; ++ns) {
                int ncol = bn + wn * 64 + ns * 16 + l16;
                v16bf bfr = *(const v16bf*)(WT + (size_t)ncol * K + kc + k0 + half * 16);
                acc[ns] = __builtin_amdgcn_wmma_f32_16x16x32_bf16(
                    false, af, false, bfr, (short)0, acc[ns], false, false);
            }
        }
    }

    #pragma unroll
    for (int ns = 0; ns < 4; ++ns) {
        int nc = bn + wn * 64 + ns * 16 + l16;
        float bval = bias ? bias[nc] : 0.0f;
        #pragma unroll
        for (int j = 0; j < 8; ++j) {
            int mr = bm + wm * 16 + half * 8 + j;
            if (mr < M) {
                float v = acc[ns][j] + bval;
                if (flags & GEMM_FLAG_GELU) v = gelu_tanh(v);
                if (resid) v += resid[(size_t)mr * Nout + nc];
                if (outF) outF[(size_t)mr * Nout + nc] = v;
                if (outB) outB[(size_t)mr * Nout + nc] = (__bf16)v;
            }
        }
    }
}

// ---------------------------------------------------------------------------
// Fused edge-LN + rel-projection + attention score. Tile: 32 edges x 256 cols.
// Raw f32 edge tile -> LDS, LN in LDS -> bf16 A, WMMA GEMM (rel in LDS only),
// score[e,h] = q[dst].(k[src]+rel)/sqrt(32). edge_features read from HBM once.
// ---------------------------------------------------------------------------
__global__ __launch_bounds__(256)
void rel_score_kernel(const float* __restrict__ EF,
                      const float* __restrict__ lng, const float* __restrict__ lnb,
                      const __bf16* __restrict__ WrelT, const float* __restrict__ brel,
                      const float* __restrict__ qbuf, const float* __restrict__ kbuf,
                      const long long* __restrict__ eidx,
                      float* __restrict__ scores, int E) {
    __shared__ __bf16 As[32][256 + 8];
    __shared__ float  fs[32][256 + 4];   // f32 staging tile, later the rel tile
    const int tid  = threadIdx.x;
    const int wid  = tid >> 5, lane = tid & 31;
    const int wm   = wid >> 2;       // 0..1
    const int wn   = wid & 3;        // 0..3
    const int half = lane >> 4, l16 = lane & 15;
    const int bm   = blockIdx.x * 32;

    // 1) cooperative load of raw f32 tile (fully coalesced float4)
    #pragma unroll
    for (int it = 0; it < 8; ++it) {
        int p = (it * 256 + tid) * 4;       // 32*256 f32 elems, 4 per thread per it
        int r = p >> 8, c = p & 255;
        float4 val = {0.0f, 0.0f, 0.0f, 0.0f};
        if (bm + r < E) val = *(const float4*)(EF + (size_t)(bm + r) * D_MODEL + c);
        *(float4*)(&fs[r][c]) = val;
    }
    __syncthreads();

    // 2) LayerNorm in LDS -> bf16 As. Each wave normalizes 4 rows.
    {
        int cbase = lane * 8;
        float4 g0 = *(const float4*)(lng + cbase), g1 = *(const float4*)(lng + cbase + 4);
        float4 b0 = *(const float4*)(lnb + cbase), b1 = *(const float4*)(lnb + cbase + 4);
        #pragma unroll
        for (int rr = 0; rr < 4; ++rr) {
            int row = wid * 4 + rr;
            const float* rp = &fs[row][0];
            float4 a0 = *(const float4*)(rp + cbase);
            float4 a1 = *(const float4*)(rp + cbase + 4);
            float s = a0.x + a0.y + a0.z + a0.w + a1.x + a1.y + a1.z + a1.w;
            float q = a0.x*a0.x + a0.y*a0.y + a0.z*a0.z + a0.w*a0.w
                    + a1.x*a1.x + a1.y*a1.y + a1.z*a1.z + a1.w*a1.w;
            #pragma unroll
            for (int m = 16; m >= 1; m >>= 1) {
                s += __shfl_xor(s, m, 32);
                q += __shfl_xor(q, m, 32);
            }
            float mean = s * (1.0f / 256.0f);
            float var  = q * (1.0f / 256.0f) - mean * mean;
            float rstd = rsqrtf(var + 1e-5f);
            v8bf o;
            o[0] = (__bf16)((a0.x - mean) * rstd * g0.x + b0.x);
            o[1] = (__bf16)((a0.y - mean) * rstd * g0.y + b0.y);
            o[2] = (__bf16)((a0.z - mean) * rstd * g0.z + b0.z);
            o[3] = (__bf16)((a0.w - mean) * rstd * g0.w + b0.w);
            o[4] = (__bf16)((a1.x - mean) * rstd * g1.x + b1.x);
            o[5] = (__bf16)((a1.y - mean) * rstd * g1.y + b1.y);
            o[6] = (__bf16)((a1.z - mean) * rstd * g1.z + b1.z);
            o[7] = (__bf16)((a1.w - mean) * rstd * g1.w + b1.w);
            *(v8bf*)(&As[row][cbase]) = o;
        }
    }
    __syncthreads();

    // 3) WMMA GEMM: rel = As @ Wrel (+brel), result parked in fs
    v8f acc[4] = {v8f{}, v8f{}, v8f{}, v8f{}};
    const __bf16* arow = &As[wm * 16 + l16][0];
    #pragma unroll
    for (int k0 = 0; k0 < 256; k0 += 32) {
        v16bf af = load_a_frag(arow, k0, half);
        #pragma unroll
        for (int ns = 0; ns < 4; ++ns) {
            int ncol = wn * 64 + ns * 16 + l16;
            v16bf bfr = *(const v16bf*)(WrelT + (size_t)ncol * D_MODEL + k0 + half * 16);
            acc[ns] = __builtin_amdgcn_wmma_f32_16x16x32_bf16(
                false, af, false, bfr, (short)0, acc[ns], false, false);
        }
    }
    #pragma unroll
    for (int ns = 0; ns < 4; ++ns) {
        int nc = wn * 64 + ns * 16 + l16;
        float bv = brel[nc];
        #pragma unroll
        for (int j = 0; j < 8; ++j)
            fs[wm * 16 + half * 8 + j][nc] = acc[ns][j] + bv;
    }
    __syncthreads();

    // 4) scores: 256 threads -> 32 edges x 8 heads
    int el = tid >> 3, h = tid & 7;
    int e = bm + el;
    if (e >= E) return;
    long long s = eidx[e];
    long long d = eidx[(size_t)E + e];
    const float* qp = qbuf + (size_t)d * D_MODEL + h * DHEAD;
    const float* kp = kbuf + (size_t)s * D_MODEL + h * DHEAD;
    float dotv = 0.0f;
    #pragma unroll
    for (int i = 0; i < DHEAD; ++i)
        dotv += qp[i] * (kp[i] + fs[el][h * DHEAD + i]);
    scores[(size_t)e * NHEAD + h] = dotv * 0.17677669529663687f;  // 1/sqrt(32)
}

// ---------------------------------------------------------------------------
// Segment max over dst via ordered-uint atomicMax
// ---------------------------------------------------------------------------
__global__ __launch_bounds__(256)
void edge_max_kernel(const float* __restrict__ scores,
                     const long long* __restrict__ eidx,
                     unsigned* __restrict__ mxu, int E) {
    int gid = blockIdx.x * 256 + threadIdx.x;
    if (gid >= E * NHEAD) return;
    int e = gid >> 3, h = gid & 7;
    long long d = eidx[(size_t)E + e];
    atomicMax(&mxu[(size_t)d * NHEAD + h], mapf(scores[gid]));
}

// ---------------------------------------------------------------------------
// p = exp(score - mx); denom += p; agg += p * v[src].
// One block per edge (uniform edge index -> scalar loads), one thread per col.
// ---------------------------------------------------------------------------
__global__ __launch_bounds__(256)
void edge_agg_kernel(const float* __restrict__ scores,
                     const long long* __restrict__ eidx,
                     const unsigned* __restrict__ mxu,
                     const float* __restrict__ vbuf,
                     float* __restrict__ denom, float* __restrict__ agg,
                     int E) {
    int e = blockIdx.x;
    if (e >= E) return;
    int c = threadIdx.x, h = c >> 5;
    long long s = eidx[e];
    long long d = eidx[(size_t)E + e];
    float sc = scores[(size_t)e * NHEAD + h];
    float m  = unmapf(mxu[(size_t)d * NHEAD + h]);
    float p  = expf(sc - m);
    atomicAdd(&agg[(size_t)d * D_MODEL + c], p * vbuf[(size_t)s * D_MODEL + c]);
    if ((c & 31) == 0) atomicAdd(&denom[(size_t)d * NHEAD + h], p);
}

// ---------------------------------------------------------------------------
// agg / (denom + 1e-9) -> bf16 (input to out-projection GEMM)
// ---------------------------------------------------------------------------
__global__ __launch_bounds__(256)
void finalize_kernel(const float* __restrict__ agg,
                     const float* __restrict__ denom,
                     __bf16* __restrict__ aggn, int Nn) {
    int gid = blockIdx.x * 256 + threadIdx.x;
    if (gid >= Nn * D_MODEL) return;
    int n = gid >> 8, h = (gid >> 5) & 7;
    aggn[gid] = (__bf16)(agg[gid] / (denom[(size_t)n * NHEAD + h] + 1e-9f));
}

// ---------------------------------------------------------------------------
// host launcher
// ---------------------------------------------------------------------------
static inline size_t alignup(size_t x) { return (x + 255) & ~(size_t)255; }

extern "C" void kernel_launch(void* const* d_in, const int* in_sizes, int n_in,
                              void* d_out, int out_size, void* d_ws, size_t ws_size,
                              hipStream_t stream) {
    const float*      scene_tokens  = (const float*)d_in[0];
    const float*      edge_features = (const float*)d_in[1];
    const long long*  edge_index    = (const long long*)d_in[2];
    const float *ln1_g = (const float*)d_in[3],  *ln1_b = (const float*)d_in[4];
    const float *lnr_g = (const float*)d_in[5],  *lnr_b = (const float*)d_in[6];
    const float *lnm_g = (const float*)d_in[7],  *lnm_b = (const float*)d_in[8];
    const float *Wq = (const float*)d_in[9],  *bq = (const float*)d_in[10];
    const float *Wk = (const float*)d_in[11], *bk = (const float*)d_in[12];
    const float *Wv = (const float*)d_in[13], *bv = (const float*)d_in[14];
    const float *Wrel = (const float*)d_in[15], *brel = (const float*)d_in[16];
    const float *Wo = (const float*)d_in[17], *bo = (const float*)d_in[18];
    const float *W1 = (const float*)d_in[19], *b1 = (const float*)d_in[20];
    const float *W2 = (const float*)d_in[21], *b2 = (const float*)d_in[22];

    const int Nn = in_sizes[0] / D_MODEL;   // 10000
    const int E  = in_sizes[1] / D_MODEL;   // 320000

    // ---- workspace carve-out ----
    char* p = (char*)d_ws;
    size_t off = 0;
    auto take = [&](size_t bytes) -> char* {
        char* r = p + off;
        off = alignup(off + bytes);
        return r;
    };
    __bf16* xn    = (__bf16*)take((size_t)Nn * D_MODEL * 2);
    float*  qb    = (float*) take((size_t)Nn * D_MODEL * 4);
    float*  kb    = (float*) take((size_t)Nn * D_MODEL * 4);
    float*  vb    = (float*) take((size_t)Nn * D_MODEL * 4);
    float*  sc    = (float*) take((size_t)E  * NHEAD   * 4);
    // zero region: mxu | denom | agg (contiguous)
    size_t zero_off = off;
    unsigned* mxu  = (unsigned*)take((size_t)Nn * NHEAD * 4);
    float*  denom  = (float*) take((size_t)Nn * NHEAD   * 4);
    float*  agg    = (float*) take((size_t)Nn * D_MODEL * 4);
    size_t zero_bytes = off - zero_off;
    __bf16* aggn  = (__bf16*)take((size_t)Nn * D_MODEL * 2);
    float*  scene = (float*) take((size_t)Nn * D_MODEL * 4);
    __bf16* sn    = (__bf16*)take((size_t)Nn * D_MODEL * 2);
    __bf16* hbuf  = (__bf16*)take((size_t)Nn * 4 * D_MODEL * 2);
    __bf16* WqT   = (__bf16*)take((size_t)D_MODEL * D_MODEL * 2);
    __bf16* WkT   = (__bf16*)take((size_t)D_MODEL * D_MODEL * 2);
    __bf16* WvT   = (__bf16*)take((size_t)D_MODEL * D_MODEL * 2);
    __bf16* WrelT = (__bf16*)take((size_t)D_MODEL * D_MODEL * 2);
    __bf16* WoT   = (__bf16*)take((size_t)D_MODEL * D_MODEL * 2);
    __bf16* W1T   = (__bf16*)take((size_t)D_MODEL * 4 * D_MODEL * 2);
    __bf16* W2T   = (__bf16*)take((size_t)4 * D_MODEL * D_MODEL * 2);
    (void)ws_size; (void)n_in; (void)out_size;

    hipMemsetAsync(p + zero_off, 0, zero_bytes, stream);

    // ---- weight conversion (transposed bf16) ----
    {
        int e256 = D_MODEL * D_MODEL;                    // 65536
        int e1k  = D_MODEL * 4 * D_MODEL;                // 262144
        w_to_bf16T_kernel<<<(e256 + 255) / 256, 256, 0, stream>>>(Wq,   WqT,   D_MODEL, D_MODEL);
        w_to_bf16T_kernel<<<(e256 + 255) / 256, 256, 0, stream>>>(Wk,   WkT,   D_MODEL, D_MODEL);
        w_to_bf16T_kernel<<<(e256 + 255) / 256, 256, 0, stream>>>(Wv,   WvT,   D_MODEL, D_MODEL);
        w_to_bf16T_kernel<<<(e256 + 255) / 256, 256, 0, stream>>>(Wrel, WrelT, D_MODEL, D_MODEL);
        w_to_bf16T_kernel<<<(e256 + 255) / 256, 256, 0, stream>>>(Wo,   WoT,   D_MODEL, D_MODEL);
        w_to_bf16T_kernel<<<(e1k  + 255) / 256, 256, 0, stream>>>(W1,   W1T,   D_MODEL, 4 * D_MODEL);
        w_to_bf16T_kernel<<<(e1k  + 255) / 256, 256, 0, stream>>>(W2,   W2T,   4 * D_MODEL, D_MODEL);
    }

    // ---- node prenorm ----
    ln_bf16_kernel<<<(Nn + 7) / 8, 256, 0, stream>>>(scene_tokens, ln1_g, ln1_b, xn, Nn);

    // ---- q/k/v projections ----
    {
        dim3 g(D_MODEL / 128, (Nn + 63) / 64);
        gemm_bf16_kernel<<<g, 256, 0, stream>>>(xn, WqT, bq, nullptr, qb, nullptr, Nn, D_MODEL, D_MODEL, 0);
        gemm_bf16_kernel<<<g, 256, 0, stream>>>(xn, WkT, bk, nullptr, kb, nullptr, Nn, D_MODEL, D_MODEL, 0);
        gemm_bf16_kernel<<<g, 256, 0, stream>>>(xn, WvT, bv, nullptr, vb, nullptr, Nn, D_MODEL, D_MODEL, 0);
    }

    // ---- fused edge-LN + rel projection + scores (single HBM pass over edges) ----
    rel_score_kernel<<<(E + 31) / 32, 256, 0, stream>>>(edge_features, lnr_g, lnr_b,
                                                        WrelT, brel, qb, kb, edge_index, sc, E);

    // ---- segment softmax + aggregation ----
    edge_max_kernel<<<(E * NHEAD + 255) / 256, 256, 0, stream>>>(sc, edge_index, mxu, E);
    edge_agg_kernel<<<E, 256, 0, stream>>>(sc, edge_index, mxu, vb, denom, agg, E);
    finalize_kernel<<<(Nn * D_MODEL + 255) / 256, 256, 0, stream>>>(agg, denom, aggn, Nn);

    // ---- out projection + residual ----
    {
        dim3 g(D_MODEL / 128, (Nn + 63) / 64);
        gemm_bf16_kernel<<<g, 256, 0, stream>>>(aggn, WoT, bo, scene_tokens, scene, nullptr,
                                                Nn, D_MODEL, D_MODEL, 0);
    }

    // ---- MLP ----
    ln_bf16_kernel<<<(Nn + 7) / 8, 256, 0, stream>>>(scene, lnm_g, lnm_b, sn, Nn);
    {
        dim3 g1(4 * D_MODEL / 128, (Nn + 63) / 64);
        gemm_bf16_kernel<<<g1, 256, 0, stream>>>(sn, W1T, b1, nullptr, nullptr, hbuf,
                                                 Nn, D_MODEL, 4 * D_MODEL, GEMM_FLAG_GELU);
        dim3 g2(D_MODEL / 128, (Nn + 63) / 64);
        gemm_bf16_kernel<<<g2, 256, 0, stream>>>(hbuf, W2T, b2, scene, (float*)d_out, nullptr,
                                                 Nn, 4 * D_MODEL, D_MODEL, 0);
    }
}
